// DimeNet_49117245997500
// MI455X (gfx1250) — matrix operand order repaired and checked by reading, not compiled, not run on hardware
//
#include <hip/hip_runtime.h>
#include <hip/hip_bf16.h>
#include <math.h>

typedef _Float16 f16;
typedef __attribute__((ext_vector_type(16))) _Float16 v16h;
typedef __attribute__((ext_vector_type(8)))  float    v8f;

#define NSPH 7
#define NRAD 6
#define PI_D 3.14159265358979323846

// ---------------------------------------------------------------------------
// Fully unguarded GEMM via v_wmma_f32_16x16x32_f16.
// Preconditions (arranged by host code): grid.x covers M16 = roundup(M,16)
// rows, grid.y = N/64, K % 32 == 0, N % 64 == 0.
// A is f32 [M16,K]; Wt is f16, TRANSPOSED [N,K].
// Each wave computes a 16(M) x 64(N) strip: A fragment loaded once per K-step,
// reused by 4 back-to-back WMMAs (independent accumulators).
// Y[M16,N] = post(A @ W): (+bias) -> (silu?) -> (*mul) -> (+res).
// ---------------------------------------------------------------------------
__global__ void gemm_f16wmma(const float* __restrict__ A, const f16* __restrict__ Wt,
                             const float* __restrict__ bias, const float* __restrict__ mul,
                             const float* __restrict__ res, float* __restrict__ Y,
                             int K, int N, int act)
{
    const int tm = blockIdx.x, tn = blockIdx.y;   // tn indexes 64-wide strips
    const int lane  = threadIdx.x & 31;
    const int lmod  = lane & 15;
    const int lhalf = lane >> 4;

    const float4* Arow = (const float4*)(A + (size_t)(tm * 16 + lmod) * K);
    const f16* Wrow0 = Wt + (size_t)(tn * 64 + 0  + lmod) * K;
    const f16* Wrow1 = Wt + (size_t)(tn * 64 + 16 + lmod) * K;
    const f16* Wrow2 = Wt + (size_t)(tn * 64 + 32 + lmod) * K;
    const f16* Wrow3 = Wt + (size_t)(tn * 64 + 48 + lmod) * K;

    v8f acc0 = {}, acc1 = {}, acc2 = {}, acc3 = {};
    for (int k0 = 0; k0 < K; k0 += 32) {
        // A lane slots: s<8 -> k = k0+8*half+s ; s>=8 -> k = k0+16+8*half+(s-8)
        const int b0 = (k0 >> 2) + 2 * lhalf;
        float4 a0 = Arow[b0];
        float4 a1 = Arow[b0 + 1];
        float4 a2 = Arow[b0 + 4];
        float4 a3 = Arow[b0 + 5];
        if (k0 + 32 < K)
            __builtin_prefetch((const void*)(Arow + b0 + 8), 0, 1);
        v16h a;
        a[0]  = (f16)a0.x; a[1]  = (f16)a0.y; a[2]  = (f16)a0.z; a[3]  = (f16)a0.w;
        a[4]  = (f16)a1.x; a[5]  = (f16)a1.y; a[6]  = (f16)a1.z; a[7]  = (f16)a1.w;
        a[8]  = (f16)a2.x; a[9]  = (f16)a2.y; a[10] = (f16)a2.z; a[11] = (f16)a2.w;
        a[12] = (f16)a3.x; a[13] = (f16)a3.y; a[14] = (f16)a3.z; a[15] = (f16)a3.w;
        // B lane slots: column n, s<8 -> k = k0+8*half+s ; s>=8 -> k = k0+16+8*half+(s-8)
        typedef __attribute__((ext_vector_type(8))) _Float16 v8h_;
        const int ko = k0 + 8 * lhalf;
#define LDB(W) __builtin_shufflevector(*(const v8h_*)((W) + ko), *(const v8h_*)((W) + ko + 16), 0,1,2,3,4,5,6,7,8,9,10,11,12,13,14,15)
        v16h b0v = LDB(Wrow0);
        v16h b1v = LDB(Wrow1);
        v16h b2v = LDB(Wrow2);
        v16h b3v = LDB(Wrow3);
#undef LDB
        acc0 = __builtin_amdgcn_wmma_f32_16x16x32_f16(false, a, false, b0v, (short)0, acc0, false, false);
        acc1 = __builtin_amdgcn_wmma_f32_16x16x32_f16(false, a, false, b1v, (short)0, acc1, false, false);
        acc2 = __builtin_amdgcn_wmma_f32_16x16x32_f16(false, a, false, b2v, (short)0, acc2, false, false);
        acc3 = __builtin_amdgcn_wmma_f32_16x16x32_f16(false, a, false, b3v, (short)0, acc3, false, false);
    }

#pragma unroll
    for (int nt = 0; nt < 4; ++nt) {
        v8f acc = (nt == 0) ? acc0 : (nt == 1) ? acc1 : (nt == 2) ? acc2 : acc3;
        const int n = tn * 64 + nt * 16 + lmod;
        const float bv = bias ? bias[n] : 0.0f;
#pragma unroll
        for (int r = 0; r < 8; ++r) {
            size_t m = (size_t)(tm * 16 + r + 8 * lhalf);
            float v = acc[r] + bv;
            if (act) v = v / (1.0f + __expf(-v));
            size_t o = m * N + n;
            if (mul) v *= mul[o];
            if (res) v += res[o];
            Y[o] = v;
        }
    }
}

// ---------------------------------------------------------------------------
// Weight prep: Wt[n*Kp + k] = f16( k<K ? W[k*N+n] : 0 )   (transpose + pad K)
// ---------------------------------------------------------------------------
__global__ void prep_wt(const float* __restrict__ W, f16* __restrict__ Wt,
                        int K, int N, int Kp, long total)
{
    long idx = (long)blockIdx.x * blockDim.x + threadIdx.x;
    if (idx >= total) return;
    int n = (int)(idx / Kp), k = (int)(idx % Kp);
    Wt[idx] = (f16)((k < K) ? W[(size_t)k * N + n] : 0.0f);
}

// ---------------------------------------------------------------------------
// Small naive matmul (tiny K): C[M,N] = A[M,K] @ B[K,N]
// ---------------------------------------------------------------------------
__global__ void small_matmul(const float* __restrict__ A, const float* __restrict__ B,
                             float* __restrict__ C, long MN, int K, int N)
{
    long idx = (long)blockIdx.x * blockDim.x + threadIdx.x;
    if (idx >= MN) return;
    long m = idx / N; int n = (int)(idx % N);
    float s = 0.0f;
    for (int k = 0; k < K; ++k) s += A[(size_t)m * K + k] * B[(size_t)k * N + n];
    C[idx] = s;
}

// ---------------------------------------------------------------------------
// Spherical Bessel zeros / normalizers / Y coefficients (1 thread, bisection).
// ---------------------------------------------------------------------------
__device__ double sph_jn_d(int l, double x)
{
    double j0 = sin(x) / x;
    if (l == 0) return j0;
    double j1 = sin(x) / (x * x) - cos(x) / x;
    double jm = j0, jc = j1;
    for (int n = 1; n < l; ++n) { double t = (2.0 * n + 1.0) / x * jc - jm; jm = jc; jc = t; }
    return jc;
}

__global__ void bessel_consts(float* __restrict__ zeros, float* __restrict__ norms,
                              float* __restrict__ ycoef)
{
    if (threadIdx.x != 0 || blockIdx.x != 0) return;
    double pts[NSPH + NRAD];
    int cnt = NSPH + NRAD;
    for (int i = 0; i < cnt; ++i) pts[i] = (i + 1) * PI_D;
    for (int r = 0; r < NRAD; ++r) zeros[r] = (float)pts[r];
    for (int l = 1; l < NSPH; ++l) {
        double np_[NSPH + NRAD];
        for (int i = 0; i < cnt - 1; ++i) {
            double lo = pts[i], hi = pts[i + 1];
            double flo = sph_jn_d(l, lo);
            for (int it = 0; it < 80; ++it) {
                double mid = 0.5 * (lo + hi);
                if (flo * sph_jn_d(l, mid) <= 0.0) hi = mid;
                else { lo = mid; flo = sph_jn_d(l, lo); }
            }
            np_[i] = 0.5 * (lo + hi);
        }
        cnt--;
        for (int i = 0; i < cnt; ++i) pts[i] = np_[i];
        for (int r = 0; r < NRAD; ++r) zeros[l * NRAD + r] = (float)pts[r];
    }
    for (int l = 0; l < NSPH; ++l)
        for (int r = 0; r < NRAD; ++r)
            norms[l * NRAD + r] =
                (float)(sqrt(2.0) / fabs(sph_jn_d(l + 1, (double)zeros[l * NRAD + r])));
    for (int l = 0; l < NSPH; ++l)
        ycoef[l] = (float)sqrt((2.0 * l + 1.0) / (4.0 * PI_D));
}

// ---------------------------------------------------------------------------
// Per-edge geometry: dist -> envelope, rbf[E,6], srad[E,42] (env * norm * j_l)
// ---------------------------------------------------------------------------
__global__ void edge_geom(const float* __restrict__ pos, const int* __restrict__ esrc,
                          const int* __restrict__ edst, const float* __restrict__ freq,
                          const float* __restrict__ zeros, const float* __restrict__ norms,
                          float* __restrict__ rbf, float* __restrict__ srad, int E)
{
    int e = blockIdx.x * blockDim.x + threadIdx.x;
    if (e >= E) return;
    int vi = edst[e], vj = esrc[e];
    float dx = pos[3 * vi] - pos[3 * vj];
    float dy = pos[3 * vi + 1] - pos[3 * vj + 1];
    float dz = pos[3 * vi + 2] - pos[3 * vj + 2];
    float du = sqrtf(dx * dx + dy * dy + dz * dz) * (1.0f / 5.0f);
    // envelope p=6: 1/x - 28 x^5 + 48 x^6 - 21 x^7
    float x2 = du * du;
    float x5 = x2 * x2 * du;
    float env = 1.0f / du - 28.0f * x5 + 48.0f * x5 * du - 21.0f * x5 * x2;
    for (int r = 0; r < NRAD; ++r)
        rbf[(size_t)e * NRAD + r] = env * sinf(freq[r] * du);
    float* sr = srad + (size_t)e * (NSPH * NRAD);
    for (int l = 0; l < NSPH; ++l) {
        for (int r = 0; r < NRAD; ++r) {
            float xx = du * zeros[l * NRAD + r];
            float j0 = sinf(xx) / xx;
            float jl;
            if (l == 0) jl = j0;
            else {
                float j1 = sinf(xx) / (xx * xx) - cosf(xx) / xx;
                float jm = j0, jc = j1;
                for (int n = 1; n < l; ++n) { float t = (2.0f * n + 1.0f) / xx * jc - jm; jm = jc; jc = t; }
                jl = jc;
            }
            sr[l * NRAD + r] = env * norms[l * NRAD + r] * jl;
        }
    }
}

// ---------------------------------------------------------------------------
// Per-triplet: angle -> Legendre cbf[7], sbf[T,64] (cols 42..63 zero-padded)
// ---------------------------------------------------------------------------
__global__ void trip_sbf(const float* __restrict__ pos, const int* __restrict__ esrc,
                         const int* __restrict__ edst, const int* __restrict__ ikj,
                         const int* __restrict__ iji, const float* __restrict__ srad,
                         const float* __restrict__ ycoef, float* __restrict__ sbf, int T)
{
    int t = blockIdx.x * blockDim.x + threadIdx.x;
    if (t >= T) return;
    int eji = iji[t], ekj = ikj[t];
    int vi = edst[eji], vj = esrc[eji], vk = esrc[ekj];
    float jx = pos[3 * vj] - pos[3 * vi], jy = pos[3 * vj + 1] - pos[3 * vi + 1], jz = pos[3 * vj + 2] - pos[3 * vi + 2];
    float kx = pos[3 * vk] - pos[3 * vj], ky = pos[3 * vk + 1] - pos[3 * vj + 1], kz = pos[3 * vk + 2] - pos[3 * vj + 2];
    float a = jx * kx + jy * ky + jz * kz;
    float cx = jy * kz - jz * ky, cy = jz * kx - jx * kz, cz = jx * ky - jy * kx;
    float b2 = cx * cx + cy * cy + cz * cz;
    float r = sqrtf(a * a + b2);
    float ct = (r > 0.0f) ? a / r : 1.0f;   // cos(atan2(|cross|, dot))
    float P[NSPH];
    P[0] = 1.0f; P[1] = ct;
    for (int l = 1; l < NSPH - 1; ++l)
        P[l + 1] = ((2.0f * l + 1.0f) * ct * P[l] - (float)l * P[l - 1]) / (float)(l + 1);
    const float* sr = srad + (size_t)ekj * (NSPH * NRAD);
    float* ob = sbf + (size_t)t * 64;
    for (int l = 0; l < NSPH; ++l) {
        float c = ycoef[l] * P[l];
        for (int q = 0; q < NRAD; ++q) ob[l * NRAD + q] = sr[l * NRAD + q] * c;
    }
    for (int q = NSPH * NRAD; q < 64; ++q) ob[q] = 0.0f;
}

// ---------------------------------------------------------------------------
// Embedding concat input A[E,384] = [ emb[z_i] | emb[z_j] | silu(rbf@Wr + br) ]
// ---------------------------------------------------------------------------
__global__ void build_acat(const int* __restrict__ z, const int* __restrict__ esrc,
                           const int* __restrict__ edst, const float* __restrict__ emb_table,
                           const float* __restrict__ rW, const float* __restrict__ rB,
                           const float* __restrict__ rbf, float* __restrict__ A, long total)
{
    long idx = (long)blockIdx.x * blockDim.x + threadIdx.x;
    if (idx >= total) return;
    int e = (int)(idx / 384), c = (int)(idx % 384);
    float v;
    if (c < 128)       v = emb_table[(size_t)z[edst[e]] * 128 + c];
    else if (c < 256)  v = emb_table[(size_t)z[esrc[e]] * 128 + (c - 128)];
    else {
        int cc = c - 256;
        float s = rB[cc];
        for (int r = 0; r < NRAD; ++r) s += rbf[(size_t)e * NRAD + r] * rW[r * 128 + cc];
        v = s / (1.0f + __expf(-s));
    }
    A[idx] = v;
}

// elem_table[z[i]] * 2^((1-scaler)^2)
__global__ void build_ebfv(const int* __restrict__ z, const int* __restrict__ edst,
                           const float* __restrict__ elem_table, const float* __restrict__ scaler,
                           float* __restrict__ out, long total)
{
    long idx = (long)blockIdx.x * blockDim.x + threadIdx.x;
    if (idx >= total) return;
    int e = (int)(idx >> 7), c = (int)(idx & 127);
    float s = 1.0f - scaler[0];
    out[idx] = elem_table[(size_t)z[edst[e]] * 128 + c] * exp2f(s * s);
}

// out block front: g = (rbf @ Wr) * x, segment-sum over dst node (atomic)
__global__ void out_rbf_scatter(const float* __restrict__ rbf, const float* __restrict__ Wr,
                                const float* __restrict__ x, const int* __restrict__ edst,
                                float* __restrict__ hn, long total)
{
    long idx = (long)blockIdx.x * blockDim.x + threadIdx.x;
    if (idx >= total) return;
    int e = (int)(idx >> 7), c = (int)(idx & 127);
    float g = 0.0f;
    for (int r = 0; r < NRAD; ++r) g += rbf[(size_t)e * NRAD + r] * Wr[r * 128 + c];
    g *= x[idx];
    atomicAdd(&hn[(size_t)edst[e] * 128 + c], g);
}

// triplet gather-scale-scatter: agg[idx_ji] += xdn[idx_kj] * sb
__global__ void trip_scatter(const float* __restrict__ xdn, const float* __restrict__ sb,
                             const int* __restrict__ ikj, const int* __restrict__ iji,
                             float* __restrict__ agg, long total)
{
    long idx = (long)blockIdx.x * blockDim.x + threadIdx.x;
    if (idx >= total) return;
    long t = idx >> 6; int c = (int)(idx & 63);
    float v = xdn[(size_t)ikj[t] * 64 + c] * sb[idx];
    atomicAdd(&agg[(size_t)iji[t] * 64 + c], v);
}

// final per-node dot with W[256] -> atomic add into scalar output
__global__ void final_dot(const float* __restrict__ H, const float* __restrict__ Wf,
                          float* __restrict__ out, int Nn, int C)
{
    int n = blockIdx.x * blockDim.x + threadIdx.x;
    if (n >= Nn) return;
    float s = 0.0f;
    for (int c = 0; c < C; ++c) s += H[(size_t)n * C + c] * Wf[c];
    atomicAdd(out, s);
}

// ---------------------------------------------------------------------------
// Host-side orchestration
// ---------------------------------------------------------------------------
extern "C" void kernel_launch(void* const* d_in, const int* in_sizes, int n_in,
                              void* d_out, int out_size, void* d_ws, size_t ws_size,
                              hipStream_t stream)
{
    (void)ws_size;
    if (n_in < 155 || !d_ws) return;

    const int*   z        = (const int*)d_in[0];
    const float* pos      = (const float*)d_in[1];
    const int*   edge_src = (const int*)d_in[2];
    const int*   edge_dst = (const int*)d_in[3];
    const int*   idx_kj   = (const int*)d_in[4];
    const int*   idx_ji   = (const int*)d_in[5];
    const int Nn  = in_sizes[0];
    const int E   = in_sizes[2];
    const int T   = in_sizes[4];
    const int E16 = (E + 15) & ~15;
    const int T16 = (T + 15) & ~15;

    // params flattened as a JAX pytree (dict keys sorted, lists in order)
    int p = 6;
    auto F = [&](int i) { return (const float*)d_in[i]; };
    const float* elem_table = F(p++);
    const float* embL_W = F(p++); const float* embL_b = F(p++);
    const float* embR_W = F(p++); const float* embR_b = F(p++);
    const float* emb_table = F(p++);

    const float *i_aW[4][2][2], *i_aB[4][2][2];
    const float *i_bW[4][2],    *i_bB[4][2];
    const float *i_down[4], *i_jiW[4], *i_jiB[4], *i_kjW[4], *i_kjB[4];
    const float *i_linW[4], *i_linB[4], *i_r1[4], *i_r2[4], *i_s1[4], *i_s2[4], *i_up[4];
    for (int b = 0; b < 4; ++b) {
        for (int l = 0; l < 2; ++l)
            for (int m = 0; m < 2; ++m) { i_aW[b][l][m] = F(p++); i_aB[b][l][m] = F(p++); }
        for (int m = 0; m < 2; ++m)     { i_bW[b][m]    = F(p++); i_bB[b][m]    = F(p++); }
        i_down[b] = F(p++);
        i_jiW[b] = F(p++); i_jiB[b] = F(p++);
        i_kjW[b] = F(p++); i_kjB[b] = F(p++);
        i_linW[b] = F(p++); i_linB[b] = F(p++);
        i_r1[b] = F(p++); i_r2[b] = F(p++); i_s1[b] = F(p++); i_s2[b] = F(p++);
        i_up[b] = F(p++);
    }
    const float *o_fin[5], *o_lW[5][3], *o_lB[5][3], *o_rbf[5], *o_up[5];
    for (int b = 0; b < 5; ++b) {
        o_fin[b] = F(p++);
        for (int l = 0; l < 3; ++l) { o_lW[b][l] = F(p++); o_lB[b][l] = F(p++); }
        o_rbf[b] = F(p++); o_up[b] = F(p++);
    }
    const float* rbf_freq = F(p++);
    const float* scaler   = F(p++);

    // workspace bump allocator (256B aligned)
    char* wp = (char*)d_ws;
    auto allocf = [&](size_t nflt) {
        float* r = (float*)wp; wp += ((nflt * 4) + 255) & ~(size_t)255; return r;
    };
    auto alloch = [&](size_t nh) {
        f16* r = (f16*)wp; wp += ((nh * 2) + 255) & ~(size_t)255; return r;
    };

    float* c_zeros = allocf(NSPH * NRAD);
    float* c_norms = allocf(NSPH * NRAD);
    float* c_ycoef = allocf(8);
    float* rbf  = allocf((size_t)E16 * NRAD);
    float* srad = allocf((size_t)E16 * NSPH * NRAD);
    float* sbf  = allocf((size_t)T16 * 64);          // K padded 42 -> 64
    float* Acat = allocf((size_t)E16 * 384);
    float* ebfv = allocf((size_t)E16 * 128);
    float* x    = allocf((size_t)E16 * 128);
    float* rb   = allocf((size_t)E16 * 128);
    float* xji  = allocf((size_t)E16 * 128);
    float* xkj  = allocf((size_t)E16 * 128);
    float* xdn  = allocf((size_t)E16 * 64);
    float* sb   = allocf((size_t)T16 * 64);
    float* agg  = allocf((size_t)E16 * 64);
    float* hbuf = allocf((size_t)E16 * 128);
    float* tbuf = allocf((size_t)E16 * 128);
    float* comb6   = allocf(6 * 128);
    float* comb42f = allocf(42 * 64);
    float* hn  = allocf((size_t)Nn * 128);
    float* hup = allocf((size_t)Nn * 256);
    float* o1  = allocf((size_t)Nn * 256);
    float* o2  = allocf((size_t)Nn * 256);

    // transposed f16 weights
    f16* w_embL = alloch((size_t)128 * 384);
    f16 *w_ji[4], *w_kj[4], *w_lin[4], *w_b[4][2], *w_a[4][2][2], *w_down[4], *w_up[4];
    for (int b = 0; b < 4; ++b) {
        w_ji[b] = alloch(128 * 128); w_kj[b] = alloch(128 * 128); w_lin[b] = alloch(128 * 128);
        for (int m = 0; m < 2; ++m) w_b[b][m] = alloch(128 * 128);
        for (int l = 0; l < 2; ++l)
            for (int m = 0; m < 2; ++m) w_a[b][l][m] = alloch(128 * 128);
        w_down[b] = alloch(64 * 128);
        w_up[b]   = alloch(128 * 64);
    }
    f16 *w_oup[5], *w_ol[5][3];
    for (int b = 0; b < 5; ++b) {
        w_oup[b] = alloch(256 * 128);
        for (int l = 0; l < 3; ++l) w_ol[b][l] = alloch(256 * 256);
    }
    f16* w_comb42 = alloch(64 * 64);

    auto GRD = [](long total) { return (unsigned)((total + 255) / 256); };
    auto PREP = [&](const float* W, f16* Wt, int K, int N, int Kp) {
        long total = (long)N * Kp;
        prep_wt<<<GRD(total), 256, 0, stream>>>(W, Wt, K, N, Kp, total);
    };
    auto GEMM = [&](const float* A, const f16* Wt, const float* bias, const float* mul,
                    const float* res, float* Y, long M16, int K, int N, int act) {
        dim3 g((unsigned)(M16 / 16), (unsigned)(N / 64));
        gemm_f16wmma<<<g, 32, 0, stream>>>(A, Wt, bias, mul, res, Y, K, N, act);
    };

    // ---- prep all transposed f16 weights ----
    PREP(embL_W, w_embL, 384, 128, 384);
    for (int b = 0; b < 4; ++b) {
        PREP(i_jiW[b],  w_ji[b],  128, 128, 128);
        PREP(i_kjW[b],  w_kj[b],  128, 128, 128);
        PREP(i_linW[b], w_lin[b], 128, 128, 128);
        for (int m = 0; m < 2; ++m) PREP(i_bW[b][m], w_b[b][m], 128, 128, 128);
        for (int l = 0; l < 2; ++l)
            for (int m = 0; m < 2; ++m) PREP(i_aW[b][l][m], w_a[b][l][m], 128, 128, 128);
        PREP(i_down[b], w_down[b], 128, 64, 128);
        PREP(i_up[b],   w_up[b],   64, 128, 64);
    }
    for (int b = 0; b < 5; ++b) {
        PREP(o_up[b], w_oup[b], 128, 256, 128);
        for (int l = 0; l < 3; ++l) PREP(o_lW[b][l], w_ol[b][l], 256, 256, 256);
    }

    // ---- basis constants + edge/triplet geometry ----
    bessel_consts<<<1, 1, 0, stream>>>(c_zeros, c_norms, c_ycoef);
    edge_geom<<<GRD(E), 256, 0, stream>>>(pos, edge_src, edge_dst, rbf_freq,
                                          c_zeros, c_norms, rbf, srad, E);
    trip_sbf<<<GRD(T), 256, 0, stream>>>(pos, edge_src, edge_dst, idx_kj, idx_ji,
                                         srad, c_ycoef, sbf, T);

    // ---- embedding block ----
    build_acat<<<GRD((long)E * 384), 256, 0, stream>>>(z, edge_src, edge_dst, emb_table,
                                                       embR_W, embR_b, rbf, Acat, (long)E * 384);
    build_ebfv<<<GRD((long)E * 128), 256, 0, stream>>>(z, edge_dst, elem_table, scaler,
                                                       ebfv, (long)E * 128);
    GEMM(Acat, w_embL, embL_b, nullptr, ebfv, x, E16, 384, 128, 1);

    hipMemsetAsync(d_out, 0, (size_t)out_size * sizeof(float), stream);

    auto OUT = [&](int b) {
        hipMemsetAsync(hn, 0, (size_t)Nn * 128 * 4, stream);
        out_rbf_scatter<<<GRD((long)E * 128), 256, 0, stream>>>(rbf, o_rbf[b], x, edge_dst,
                                                                hn, (long)E * 128);
        GEMM(hn, w_oup[b], nullptr, nullptr, nullptr, hup, Nn, 128, 256, 0);
        GEMM(hup, w_ol[b][0], o_lB[b][0], nullptr, nullptr, o1, Nn, 256, 256, 1);
        GEMM(o1,  w_ol[b][1], o_lB[b][1], nullptr, nullptr, o2, Nn, 256, 256, 1);
        GEMM(o2,  w_ol[b][2], o_lB[b][2], nullptr, nullptr, o1, Nn, 256, 256, 1);
        final_dot<<<GRD(Nn), 256, 0, stream>>>(o1, o_fin[b], (float*)d_out, Nn, 256);
    };

    auto INT = [&](int b) {
        // collapse linear basis projections: comb6 = rbf1@rbf2, comb42 = sbf1@sbf2
        small_matmul<<<GRD(6 * 128), 256, 0, stream>>>(i_r1[b], i_r2[b], comb6, 6 * 128, 8, 128);
        small_matmul<<<GRD(42 * 64), 256, 0, stream>>>(i_s1[b], i_s2[b], comb42f, 42 * 64, 8, 64);
        PREP(comb42f, w_comb42, 42, 64, 64);
        small_matmul<<<GRD((long)E * 128), 256, 0, stream>>>(rbf, comb6, rb, (long)E * 128, 6, 128);
        GEMM(sbf, w_comb42, nullptr, nullptr, nullptr, sb, T16, 64, 64, 0);    // big: T x 64 x 64
        GEMM(x, w_ji[b], i_jiB[b], nullptr, nullptr, xji, E16, 128, 128, 1);
        GEMM(x, w_kj[b], i_kjB[b], rb, nullptr, xkj, E16, 128, 128, 1);        // silu(.)*rb
        GEMM(xkj, w_down[b], nullptr, nullptr, nullptr, xdn, E16, 128, 64, 1);
        hipMemsetAsync(agg, 0, (size_t)E16 * 64 * 4, stream);
        trip_scatter<<<GRD((long)T * 64), 256, 0, stream>>>(xdn, sb, idx_kj, idx_ji,
                                                            agg, (long)T * 64);
        GEMM(agg, w_up[b], nullptr, nullptr, xji, hbuf, E16, 64, 128, 1);      // h = xji + silu(agg@up)
        // before residual layer
        GEMM(hbuf, w_b[b][0], i_bB[b][0], nullptr, nullptr, tbuf, E16, 128, 128, 1);
        GEMM(tbuf, w_b[b][1], i_bB[b][1], nullptr, hbuf, hbuf, E16, 128, 128, 1);
        // h = silu(h@lin + b) + x  -> new x
        GEMM(hbuf, w_lin[b], i_linB[b], nullptr, x, x, E16, 128, 128, 1);
        // after residual layers
        for (int l = 0; l < 2; ++l) {
            GEMM(x, w_a[b][l][0], i_aB[b][l][0], nullptr, nullptr, tbuf, E16, 128, 128, 1);
            GEMM(tbuf, w_a[b][l][1], i_aB[b][l][1], nullptr, x, x, E16, 128, 128, 1);
        }
    };

    OUT(0);
    for (int b = 0; b < 4; ++b) { INT(b); OUT(b + 1); }
}
